// Decoder_55654186222328
// MI455X (gfx1250) — matrix operand order repaired and checked by reading, not compiled, not run on hardware
//
#include <hip/hip_runtime.h>
#include <hip/hip_bf16.h>
#include <cstdint>

// CDNA5 / gfx1250: wave32, WMMA 16x16x32 bf16
typedef __attribute__((ext_vector_type(16))) __bf16 v16bf;
typedef __attribute__((ext_vector_type(8)))  __bf16 v8bf;
typedef __attribute__((ext_vector_type(8)))  float  v8f;

#define HB 128   // hidden size
#define MB 512   // map tokens
#define AB 64    // agents total
#define AT 32    // agents per block
#define RT (AT/16)
#define NW 8     // waves per block
#define HEADS 8
#define HD 16

__device__ __forceinline__ v8f wmma_bf16f(v16bf a, v16bf b, v8f c) {
  return __builtin_amdgcn_wmma_f32_16x16x32_bf16(false, a, false, b, (short)0, c, false, false);
}

union frag16 { v16bf v; v8bf h[2]; };

// A fragment (16x32 bf16) from row-major bf16 src (16B aligned rows/k-steps):
// lane r=lane&15 holds row; elems i => K=hi*16+i (16 contiguous bf16)
__device__ __forceinline__ v16bf load_a_bf(const __bf16* src, int ld, int lane, int row0, int k0) {
  int r = lane & 15, hi = (lane >> 4) & 1;
  const __bf16* p = src + (size_t)(row0 + r) * ld + k0 + hi * 16;
  frag16 u;
  u.h[0] = *(const v8bf*)p;
  u.h[1] = *(const v8bf*)(p + 8);
  return u.v;
}

// B fragment (32x16) where B[k][n] = W[n0+n][k0+k], W row-major bf16 [N][ldw]
// lane n=lane&15 holds column; elems i => K = (lane>>4)*16 + i  (16 contiguous bf16)
__device__ __forceinline__ v16bf load_b_bf(const __bf16* W, int ldw, int n0, int k0, int lane) {
  int n = lane & 15;
  const __bf16* p = W + (size_t)(n0 + n) * ldw + k0 + ((lane >> 4) & 1) * 16;
  frag16 u;
  u.h[0] = *(const v8bf*)p;
  u.h[1] = *(const v8bf*)(p + 8);
  return u.v;
}

// C(AT x 16*nct) = A(bf16) @ W^T + bias(f32), optional relu, store bf16
template<int KDIM, bool RELU>
__device__ __forceinline__ void gemm_aw(const __bf16* As, int lda,
                                        const __bf16* W, const float* bias,
                                        __bf16* Cs, int ldc, int nct,
                                        int wave, int lane) {
  int r = lane & 15, hi = (lane >> 4) & 1;
  int nt = RT * nct;
  for (int t = wave; t < nt; t += NW) {   // wave-uniform
    int rt = t % RT, ct = t / RT;
    int a0 = rt * 16, n0 = ct * 16;
    v8f acc = {};
#pragma unroll
    for (int k0 = 0; k0 < KDIM; k0 += 32)
      acc = wmma_bf16f(load_a_bf(As, lda, lane, a0, k0),
                       load_b_bf(W, KDIM, n0, k0, lane), acc);
    float bb = bias[n0 + r];
#pragma unroll
    for (int i = 0; i < 8; ++i) {
      float v = acc[i] + bb;
      if (RELU) v = fmaxf(v, 0.f);
      Cs[(size_t)(a0 + hi * 8 + i) * ldc + n0 + r] = (__bf16)v;
    }
  }
}

// ---- precompute: kh[b][m][d], vhT[b][d][m] bf16 (time-invariant; 33.5MB, L2-resident)
__global__ __launch_bounds__(256) void precompute_kv(
    const float* __restrict__ me, const float* __restrict__ wk, const float* __restrict__ bk,
    const float* __restrict__ wv, const float* __restrict__ bv,
    __bf16* __restrict__ khg, __bf16* __restrict__ vhT)
{
  int b = blockIdx.x;
  for (int idx = threadIdx.x; idx < MB * HB; idx += blockDim.x) {
    int m = idx >> 7, d = idx & 127;
    const float* x   = me + ((size_t)b * MB + m) * HB;
    const float* wkr = wk + (size_t)d * HB;
    const float* wvr = wv + (size_t)d * HB;
    float ak = bk[d], av = bv[d];
#pragma unroll 4
    for (int k = 0; k < HB; ++k) { float xv = x[k]; ak += xv * wkr[k]; av += xv * wvr[k]; }
    khg[((size_t)b * MB + m) * HB + d] = (__bf16)ak;
    vhT[((size_t)b * HB + d) * MB + m] = (__bf16)av;
  }
}

// ---- one-shot f32 -> bf16 weight conversion
__global__ __launch_bounds__(256) void cvt_bf16(const float* __restrict__ src,
                                                __bf16* __restrict__ dst, int n) {
  int i = blockIdx.x * blockDim.x + threadIdx.x;
  if (i < n) dst[i] = (__bf16)src[i];
}

__global__ __launch_bounds__(256) void decoder_kernel(
    const float* __restrict__ feat, const unsigned char* __restrict__ cur_av,
    const float* __restrict__ hidden, const float* __restrict__ context,
    const unsigned char* __restrict__ map_av, const int* __restrict__ n_ts,
    const float* __restrict__ ew1, const float* __restrict__ eb1,
    const __bf16* __restrict__ w2b, const float* __restrict__ eb2,
    const __bf16* __restrict__ w3b, const float* __restrict__ eb3,
    const __bf16* __restrict__ wqb, const float* __restrict__ bq,
    const __bf16* __restrict__ wob, const float* __restrict__ bo,
    const __bf16* __restrict__ wihb, const __bf16* __restrict__ whhb,
    const float* __restrict__ b_ih, const float* __restrict__ b_hh,
    const float* __restrict__ lin_w, const float* __restrict__ lin_b,
    const __bf16* __restrict__ khg, const __bf16* __restrict__ vhT,
    float* __restrict__ out)
{
  __shared__ __align__(16) __bf16 bufA[AT * HB];      // e1 / e3 / lstm-x
  __shared__ __align__(16) __bf16 bufB[AT * HB];      // e2 / q
  __shared__ __align__(16) __bf16 bufC[AT * HB];      // attention out (pre-wo)
  __shared__ __align__(16) __bf16 sH[AT * HB];        // h (bf16 for GEMM)
  __shared__ __align__(16) float  sC[AT * HB];        // c
  __shared__ __align__(16) __bf16 sP[NW * 16 * 32];   // per-wave P transpose scratch
  __shared__ float  sState[AT * 2];

  const int tid  = threadIdx.x;
  const int lane = tid & 31;
  const int wave = tid >> 5;
  const int r    = lane & 15;
  const int hi   = (lane >> 4) & 1;

  const int bid = blockIdx.x;
  const int b   = bid >> 1;
  const int ag0 = (bid & 1) * AT;

  const int T = *n_ts;

  for (int idx = tid; idx < AT * HB; idx += 256) {
    int a = idx >> 7, d = idx & 127;
    size_t g = ((size_t)b * AB + ag0 + a) * HB + d;
    sH[idx] = (__bf16)hidden[g];
    sC[idx] = context[g];
  }
  for (int idx = tid; idx < AT * 2; idx += 256)
    sState[idx] = feat[(size_t)(b * AB + ag0) * 2 + idx];
  __syncthreads();

  for (int t = 0; t < T; ++t) {
    // ---- MLP layer 1 (K=2, VALU): relu(state @ w1^T + b1) -> bufA
    for (int idx = tid; idx < AT * HB; idx += 256) {
      int a = idx >> 7, d = idx & 127;
      float v = sState[a * 2] * ew1[d * 2] + sState[a * 2 + 1] * ew1[d * 2 + 1] + eb1[d];
      bufA[idx] = (__bf16)fmaxf(v, 0.f);
    }
    __syncthreads();
    gemm_aw<128, true >(bufA, HB, w2b, eb2, bufB, 64, 4, wave, lane);   // e2 (AT,64)
    __syncthreads();
    gemm_aw<64,  false>(bufB, 64, w3b, eb3, bufA, HB, 8, wave, lane);   // e3 (AT,128)
    __syncthreads();
    gemm_aw<128, false>(bufA, HB, wqb, bq,  bufB, HB, 8, wave, lane);   // q  (AT,128)
    __syncthreads();

    // ---- flash attention: one head per wave, online softmax over M=512
    {
      const int head = wave;
      __bf16* sPw = sP + head * 16 * 32;
      for (int rt = 0; rt < RT; ++rt) {
        int a0 = rt * 16;
        v16bf aq;   // Q fragment: d in [head*16,head*16+16), K 16..31 zero-padded
        {
          frag16 u; v8bf zh = {};
          const __bf16* qp = bufB + (size_t)(a0 + r) * HB + head * HD;
          if (hi == 0) { u.h[0] = *(const v8bf*)qp; u.h[1] = *(const v8bf*)(qp + 8); }
          else { u.h[0] = zh; u.h[1] = zh; }
          aq = u.v;
        }
        v8f O = {};
        float mrow[8], lrow[8];
#pragma unroll
        for (int i = 0; i < 8; ++i) { mrow[i] = -1e30f; lrow[i] = 0.f; }

        for (int mt = 0; mt < MB; mt += 32) {
          v16bf bk0 = {}, bk1 = {};      // K^T fragments, K>=16 zero-padded
          const __bf16* kp = khg + ((size_t)b * MB + mt) * HB + head * HD;
          if (hi == 0) {
            frag16 u0, u1;
            const __bf16* p0 = kp + (size_t)r * HB;
            const __bf16* p1 = kp + (size_t)(16 + r) * HB;
            u0.h[0] = *(const v8bf*)p0;  u0.h[1] = *(const v8bf*)(p0 + 8);
            u1.h[0] = *(const v8bf*)p1;  u1.h[1] = *(const v8bf*)(p1 + 8);
            bk0 = u0.v; bk1 = u1.v;
          }
          if (mt + 32 < MB) {            // hint next K tile toward L0/L2
            __builtin_prefetch(kp + (size_t)(32 + r) * HB, 0, 3);
          }
          v8f z = {};
          v8f S0 = wmma_bf16f(aq, bk0, z);
          v8f S1 = wmma_bf16f(aq, bk1, z);

          bool av0 = map_av[(size_t)b * MB + mt + r]      != 0;
          bool av1 = map_av[(size_t)b * MB + mt + 16 + r] != 0;
          float t0[8], t1[8], tm[8];
#pragma unroll
          for (int i = 0; i < 8; ++i) {
            t0[i] = av0 ? S0[i] * 0.25f : -1e30f;
            t1[i] = av1 ? S1[i] * 0.25f : -1e30f;
            tm[i] = fmaxf(t0[i], t1[i]);
          }
#pragma unroll
          for (int off = 1; off < 16; off <<= 1)
#pragma unroll
            for (int i = 0; i < 8; ++i)
              tm[i] = fmaxf(tm[i], __shfl_xor(tm[i], off, 32));
          float rs[8];
#pragma unroll
          for (int i = 0; i < 8; ++i) {
            float nm = fmaxf(mrow[i], tm[i]);
            float al = __expf(mrow[i] - nm);
            float p0 = __expf(t0[i] - nm);
            float p1 = __expf(t1[i] - nm);
            mrow[i] = nm;
            O[i]    *= al;
            lrow[i] *= al;
            rs[i] = p0 + p1;
            sPw[(hi * 8 + i) * 32 + r]      = (__bf16)p0;   // C-layout -> LDS
            sPw[(hi * 8 + i) * 32 + 16 + r] = (__bf16)p1;
          }
#pragma unroll
          for (int off = 1; off < 16; off <<= 1)
#pragma unroll
            for (int i = 0; i < 8; ++i)
              rs[i] += __shfl_xor(rs[i], off, 32);
#pragma unroll
          for (int i = 0; i < 8; ++i) lrow[i] += rs[i];

          v16bf ap = load_a_bf(sPw, 32, lane, 0, 0);        // reload as A-layout
          frag16 uv;                                         // V fragment from vhT
          const __bf16* vp = vhT + ((size_t)b * HB + head * HD + r) * MB + mt + hi * 16;
          uv.h[0] = *(const v8bf*)vp;  uv.h[1] = *(const v8bf*)(vp + 8);
          O = wmma_bf16f(ap, uv.v, O);
        }
#pragma unroll
        for (int i = 0; i < 8; ++i)
          bufC[(size_t)(a0 + hi * 8 + i) * HB + head * HD + r] = (__bf16)(O[i] / lrow[i]);
      }
    }
    __syncthreads();

    gemm_aw<128, false>(bufC, HB, wob, bo, bufA, HB, 8, wave, lane);   // x = attn @ wo^T
    __syncthreads();

    // ---- LSTM: wave owns hidden cols [wave*16,wave*16+16); gate tiles in registers
    {
      const int d0 = wave * 16;
      v8f acc[RT][4] = {};
#pragma unroll
      for (int k0 = 0; k0 < HB; k0 += 32) {
        v16bf bw[4];
#pragma unroll
        for (int q = 0; q < 4; ++q) bw[q] = load_b_bf(wihb, HB, q * HB + d0, k0, lane);
#pragma unroll
        for (int rt = 0; rt < RT; ++rt) {
          v16bf ax = load_a_bf(bufA, HB, lane, rt * 16, k0);
#pragma unroll
          for (int q = 0; q < 4; ++q) acc[rt][q] = wmma_bf16f(ax, bw[q], acc[rt][q]);
        }
      }
#pragma unroll
      for (int k0 = 0; k0 < HB; k0 += 32) {
        v16bf bw[4];
#pragma unroll
        for (int q = 0; q < 4; ++q) bw[q] = load_b_bf(whhb, HB, q * HB + d0, k0, lane);
#pragma unroll
        for (int rt = 0; rt < RT; ++rt) {
          v16bf ah = load_a_bf(sH, HB, lane, rt * 16, k0);
#pragma unroll
          for (int q = 0; q < 4; ++q) acc[rt][q] = wmma_bf16f(ah, bw[q], acc[rt][q]);
        }
      }
      __syncthreads();   // all waves done reading sH before writes
      const int col = d0 + r;
      float bI  = b_ih[0 * HB + col] + b_hh[0 * HB + col];
      float bF  = b_ih[1 * HB + col] + b_hh[1 * HB + col];
      float bG  = b_ih[2 * HB + col] + b_hh[2 * HB + col];
      float bOg = b_ih[3 * HB + col] + b_hh[3 * HB + col];
#pragma unroll
      for (int rt = 0; rt < RT; ++rt) {
#pragma unroll
        for (int i = 0; i < 8; ++i) {
          int row = rt * 16 + hi * 8 + i;
          float iv = 1.f / (1.f + __expf(-(acc[rt][0][i] + bI)));
          float fv = 1.f / (1.f + __expf(-(acc[rt][1][i] + bF)));
          float gv = tanhf(acc[rt][2][i] + bG);
          float ov = 1.f / (1.f + __expf(-(acc[rt][3][i] + bOg)));
          float cold = sC[row * HB + col];
          float cn = fv * cold + iv * gv;
          float hn = ov * tanhf(cn);
          bool av = cur_av[(size_t)b * AB + ag0 + row] != 0;
          float hold = (float)sH[row * HB + col];
          sC[row * HB + col] = av ? cn : cold;
          sH[row * HB + col] = (__bf16)(av ? hn : hold);
        }
      }
    }
    __syncthreads();

    // ---- state += h @ lin_w^T + lin_b ; emit output (B,A,T,2)
    for (int idx = tid; idx < AT * 2; idx += 256) {
      int a = idx >> 1, o = idx & 1;
      const float* lw = lin_w + o * HB;
      float accv = lin_b[o];
#pragma unroll 4
      for (int d = 0; d < HB; ++d) accv += (float)sH[a * HB + d] * lw[d];
      float s = sState[idx] + accv;
      sState[idx] = s;
      out[(((size_t)b * AB + ag0 + a) * T + t) * 2 + o] = s;
    }
    __syncthreads();
  }
}

extern "C" void kernel_launch(void* const* d_in, const int* in_sizes, int n_in,
                              void* d_out, int out_size, void* d_ws, size_t ws_size,
                              hipStream_t stream) {
  const float* feat    = (const float*)d_in[0];
  const unsigned char* cav = (const unsigned char*)d_in[1];
  const float* hidden  = (const float*)d_in[2];
  const float* context = (const float*)d_in[3];
  const float* me      = (const float*)d_in[4];
  const unsigned char* mav = (const unsigned char*)d_in[5];
  const int*   nts     = (const int*)d_in[6];
  const float* ew1 = (const float*)d_in[7];
  const float* eb1 = (const float*)d_in[8];
  const float* ew2 = (const float*)d_in[9];
  const float* eb2 = (const float*)d_in[10];
  const float* ew3 = (const float*)d_in[11];
  const float* eb3 = (const float*)d_in[12];
  const float* wq  = (const float*)d_in[13];
  const float* bq  = (const float*)d_in[14];
  const float* wk  = (const float*)d_in[15];
  const float* bk  = (const float*)d_in[16];
  const float* wv  = (const float*)d_in[17];
  const float* bv  = (const float*)d_in[18];
  const float* wo  = (const float*)d_in[19];
  const float* bo  = (const float*)d_in[20];
  const float* w_ih = (const float*)d_in[21];
  const float* w_hh = (const float*)d_in[22];
  const float* b_ih = (const float*)d_in[23];
  const float* b_hh = (const float*)d_in[24];
  const float* lin_w = (const float*)d_in[25];
  const float* lin_b = (const float*)d_in[26];

  // workspace layout (bf16): khg | vhT | w2b | w3b | wqb | wob | wihb | whhb
  __bf16* khg  = (__bf16*)d_ws;
  __bf16* vhT  = khg  + (size_t)128 * MB * HB;
  __bf16* w2b  = vhT  + (size_t)128 * MB * HB;
  __bf16* w3b  = w2b  + 64 * 128;
  __bf16* wqb  = w3b  + 128 * 64;
  __bf16* wob  = wqb  + 128 * 128;
  __bf16* wihb = wob  + 128 * 128;
  __bf16* whhb = wihb + 512 * 128;

  cvt_bf16<<<(64 * 128 + 255) / 256, 256, 0, stream>>>(ew2, w2b, 64 * 128);
  cvt_bf16<<<(128 * 64 + 255) / 256, 256, 0, stream>>>(ew3, w3b, 128 * 64);
  cvt_bf16<<<(128 * 128 + 255) / 256, 256, 0, stream>>>(wq, wqb, 128 * 128);
  cvt_bf16<<<(128 * 128 + 255) / 256, 256, 0, stream>>>(wo, wob, 128 * 128);
  cvt_bf16<<<(512 * 128 + 255) / 256, 256, 0, stream>>>(w_ih, wihb, 512 * 128);
  cvt_bf16<<<(512 * 128 + 255) / 256, 256, 0, stream>>>(w_hh, whhb, 512 * 128);
  precompute_kv<<<128, 256, 0, stream>>>(me, wk, bk, wv, bv, khg, vhT);

  decoder_kernel<<<256, 256, 0, stream>>>(feat, cav, hidden, context, mav, nts,
      ew1, eb1, w2b, eb2, w3b, eb3, wqb, bq, wob, bo,
      wihb, whhb, b_ih, b_hh, lin_w, lin_b, khg, vhT, (float*)d_out);
}
